// InternImage_48747878810047
// MI455X (gfx1250) — compile-verified, matrix operand order checked by reading, not run on hardware
//
#include <hip/hip_runtime.h>
#include <hip/hip_bf16.h>

typedef __attribute__((ext_vector_type(16))) __bf16 v16bf;
typedef __attribute__((ext_vector_type(8)))  __bf16 v8bf;
typedef __attribute__((ext_vector_type(8)))  float  v8f;

#define C_   192
#define G_   12
#define K_   9
#define GC_  16
#define B_   8
#define H_   64
#define W_   64
#define NPIX (B_*H_*W_)    /* 32768 */
#define NOM  336           /* 216 (off) + 108 (mask) padded to 21*16 */
#define BROW 200           /* LDS row stride in bf16 elems: 400B, 16B-aligned, */
                           /* 100 DW -> gcd(100,64)=4 -> 16 conflict-free lanes */

static __device__ __forceinline__ __bf16 f2bf(float f) {
  union { float f; unsigned u; } cv; cv.f = f;
  unsigned u = cv.u + 0x7FFFu + ((cv.u >> 16) & 1u);   // round-to-nearest-even
  union { unsigned short s; __bf16 b; } o; o.s = (unsigned short)(u >> 16);
  return o.b;
}

// ---------------------------------------------------------------------------
// K0: convert + transpose weights to bf16 [N][K] layout; build padded om bias.
// ---------------------------------------------------------------------------
__global__ void prep_weights(const float* __restrict__ w_in,  const float* __restrict__ w_off,
                             const float* __restrict__ w_mask,const float* __restrict__ w_out,
                             const float* __restrict__ b_off, const float* __restrict__ b_mask,
                             __bf16* __restrict__ w_inT, __bf16* __restrict__ w_omT,
                             __bf16* __restrict__ w_outT, float* __restrict__ bias_om) {
  int idx = blockIdx.x * blockDim.x + threadIdx.x;
  if (idx >= NOM * C_) return;
  int n = idx / C_, k = idx - n * C_;
  float om;
  if (n < 216)      om = w_off[k * 216 + n];
  else if (n < 324) om = w_mask[k * 108 + (n - 216)];
  else              om = 0.f;
  w_omT[n * C_ + k] = f2bf(om);
  if (n < C_) {
    w_inT [n * C_ + k] = f2bf(w_in [k * C_ + n]);
    w_outT[n * C_ + k] = f2bf(w_out[k * C_ + n]);
  }
  if (k == 0) bias_om[n] = (n < 216) ? b_off[n] : (n < 324 ? b_mask[n - 216] : 0.f);
}

// ---------------------------------------------------------------------------
// K1: depthwise 3x3 (SAME, zero pad) + LayerNorm(C) + exact GELU -> x1 (bf16)
//     also emits bf16 copy of x for the w_in GEMM. One block = one pixel.
// ---------------------------------------------------------------------------
__global__ void dw_ln_gelu(const float* __restrict__ x, const float* __restrict__ dw_w,
                           const float* __restrict__ dw_b, const float* __restrict__ ln_g,
                           const float* __restrict__ ln_b,
                           __bf16* __restrict__ x1_bf, __bf16* __restrict__ x_bf) {
  __shared__ float s1[256], s2[256];
  int p = blockIdx.x;          // pixel
  int c = threadIdx.x;         // channel (blockDim == 192)
  int b = p >> 12, hw = p & 4095, h = hw >> 6, w = hw & 63;
  float acc = dw_b[c];
  #pragma unroll
  for (int t = 0; t < 9; ++t) {
    int yy = h + t / 3 - 1, xx = w + t % 3 - 1;
    if (yy >= 0 && yy < H_ && xx >= 0 && xx < W_)
      acc += x[((size_t)((b * H_ + yy) * W_ + xx)) * C_ + c] * dw_w[t * C_ + c];
  }
  x_bf[(size_t)p * C_ + c] = f2bf(x[(size_t)p * C_ + c]);
  s1[c] = acc; s2[c] = acc * acc;
  if (c < 64) { s1[C_ + c] = 0.f; s2[C_ + c] = 0.f; }
  __syncthreads();
  for (int st = 128; st > 0; st >>= 1) {
    if (c < st) { s1[c] += s1[c + st]; s2[c] += s2[c + st]; }
    __syncthreads();
  }
  float mu  = s1[0] * (1.0f / C_);
  float var = s2[0] * (1.0f / C_) - mu * mu;
  float xn  = (acc - mu) * rsqrtf(var + 1e-6f) * ln_g[c] + ln_b[c];
  float ge  = 0.5f * xn * (1.0f + erff(xn * 0.70710678118654752f));
  x1_bf[(size_t)p * C_ + c] = f2bf(ge);
}

// ---------------------------------------------------------------------------
// K2: bf16 WMMA GEMM with A-register reuse across NT N-tiles and B staged in
//     LDS via CDNA5 async copy (global_load_async_to_lds_b128 / ASYNCcnt).
//     Block = 8 waves = 8 consecutive M-tiles, all sharing one N-group.
//     grid.x = 256 * ngroups;  mt = (bx&255)*8+wave;  ng = bx>>8.
// ---------------------------------------------------------------------------
template <int NT>
__global__ void gemm_bf16(const __bf16* __restrict__ A, const __bf16* __restrict__ BT,
                          const float* __restrict__ bias, float* __restrict__ out,
                          int ncols) {
  constexpr int ROWS = NT * 16;
  __shared__ __bf16 Bs[ROWS * BROW];

  int wave = threadIdx.x >> 5;
  int lane = threadIdx.x & 31;
  int mt = (blockIdx.x & 255) * 8 + wave;   // 2048 m-tiles / 8 per block
  int ng = blockIdx.x >> 8;

  // --- stage B panel (ROWS x 192 bf16) into LDS with async copies ---------
  {
    const __bf16* bsrc = BT + (size_t)ng * ROWS * C_;
    unsigned lds_base = (unsigned)(reinterpret_cast<uintptr_t>(&Bs[0]));
    for (int i = threadIdx.x; i < ROWS * 24; i += 256) {   // 24 x 16B per row
      int row = i / 24, col = i - row * 24;
      unsigned lds = lds_base + row * (BROW * 2) + col * 16;
      const __bf16* src = bsrc + row * C_ + col * 8;
      asm volatile("global_load_async_to_lds_b128 %0, %1, off"
                   :: "v"(lds), "v"(src) : "memory");
    }
    asm volatile("s_wait_asynccnt 0x0" ::: "memory");
    __syncthreads();
  }

  int half = lane >> 4;        // 0: k 0-7,16-23; 1: k 8-15,24-31
  int l15  = lane & 15;
  const __bf16* aptr = A + (size_t)(mt * 16 + l15) * C_ + half * 8;

  v8f acc[NT];
  #pragma unroll
  for (int j = 0; j < NT; ++j) acc[j] = (v8f){};

  #pragma unroll
  for (int ks = 0; ks < 6; ++ks) {
    v8bf alo = *reinterpret_cast<const v8bf*>(aptr + ks * 32);
    v8bf ahi = *reinterpret_cast<const v8bf*>(aptr + ks * 32 + 16);
    v16bf av;
    #pragma unroll
    for (int i = 0; i < 8; ++i) { av[i] = alo[i]; av[i + 8] = ahi[i]; }
    #pragma unroll
    for (int j = 0; j < NT; ++j) {
      const __bf16* bp = &Bs[(j * 16 + l15) * BROW + ks * 32 + half * 8];
      v8bf blo = *reinterpret_cast<const v8bf*>(bp);
      v8bf bhi = *reinterpret_cast<const v8bf*>(bp + 16);
      v16bf bv;
      #pragma unroll
      for (int i = 0; i < 8; ++i) { bv[i] = blo[i]; bv[i + 8] = bhi[i]; }
      acc[j] = __builtin_amdgcn_wmma_f32_16x16x32_bf16(false, av, false, bv,
                                                       (short)0, acc[j], false, false);
    }
  }

  int rowbase = mt * 16 + half * 8;    // C/D layout: lanes16-31 hold M=8..15
  #pragma unroll
  for (int j = 0; j < NT; ++j) {
    int colg = (ng * NT + j) * 16 + l15;
    float bz = bias[colg];
    #pragma unroll
    for (int r = 0; r < 8; ++r)
      out[(size_t)(rowbase + r) * ncols + colg] = acc[j][r] + bz;
  }
}

// ---------------------------------------------------------------------------
// K4: per-(pixel,group): softmax over 9 taps + bilinear deformable gather.
//     16 lanes = 16 group channels -> 64B coalesced gathers from xp.
// ---------------------------------------------------------------------------
__global__ void deform_sample(const float* __restrict__ xp, const float* __restrict__ om,
                              __bf16* __restrict__ y_bf) {
  int pair = blockIdx.x * 16 + (threadIdx.x >> 4);   // (pixel, group)
  int lane = threadIdx.x & 15;
  int p = pair / G_;
  int g = pair - p * G_;
  int b = p >> 12, hw = p & 4095, h = hw >> 6, w = hw & 63;
  const float* omp = om + (size_t)p * NOM;

  float ml[K_];
  float mx = -1e30f;
  #pragma unroll
  for (int k = 0; k < K_; ++k) { ml[k] = omp[216 + g * K_ + k]; mx = fmaxf(mx, ml[k]); }
  float se = 0.f;
  #pragma unroll
  for (int k = 0; k < K_; ++k) { ml[k] = expf(ml[k] - mx); se += ml[k]; }
  float inv = 1.0f / se;

  int ch = g * GC_ + lane;
  float acc = 0.f;
  #pragma unroll
  for (int k = 0; k < K_; ++k) {
    float dx = omp[(g * K_ + k) * 2 + 0];
    float dy = omp[(g * K_ + k) * 2 + 1];
    float px = (float)(w + (k % 3) - 1) + dx;
    float py = (float)(h + (k / 3) - 1) + dy;
    float x0f = floorf(px), y0f = floorf(py);
    int   x0 = (int)x0f,   y0 = (int)y0f;
    float fx = px - x0f,   fy = py - y0f;
    float mk  = ml[k] * inv;
    float w00 = (1.f - fx) * (1.f - fy) * mk;
    float w10 = fx * (1.f - fy) * mk;
    float w01 = (1.f - fx) * fy * mk;
    float w11 = fx * fy * mk;
    bool xin0 = (x0 >= 0 && x0 < W_), xin1 = (x0 + 1 >= 0 && x0 + 1 < W_);
    if (y0 >= 0 && y0 < H_) {
      const float* row = xp + (size_t)((b * H_ + y0) * W_) * C_;
      if (xin0) acc += w00 * row[(size_t)x0 * C_ + ch];
      if (xin1) acc += w10 * row[(size_t)(x0 + 1) * C_ + ch];
    }
    if (y0 + 1 >= 0 && y0 + 1 < H_) {
      const float* row = xp + (size_t)((b * H_ + y0 + 1) * W_) * C_;
      if (xin0) acc += w01 * row[(size_t)x0 * C_ + ch];
      if (xin1) acc += w11 * row[(size_t)(x0 + 1) * C_ + ch];
    }
  }
  y_bf[(size_t)p * C_ + ch] = f2bf(acc);
}

// ---------------------------------------------------------------------------
extern "C" void kernel_launch(void* const* d_in, const int* in_sizes, int n_in,
                              void* d_out, int out_size, void* d_ws, size_t ws_size,
                              hipStream_t stream) {
  (void)in_sizes; (void)n_in; (void)out_size; (void)ws_size;
  const float* x      = (const float*)d_in[0];
  const float* w_in   = (const float*)d_in[1];
  const float* b_in   = (const float*)d_in[2];
  const float* dw_w   = (const float*)d_in[3];
  const float* dw_b   = (const float*)d_in[4];
  const float* ln_g   = (const float*)d_in[5];
  const float* ln_b   = (const float*)d_in[6];
  const float* w_off  = (const float*)d_in[7];
  const float* b_off  = (const float*)d_in[8];
  const float* w_mask = (const float*)d_in[9];
  const float* b_mask = (const float*)d_in[10];
  const float* w_out  = (const float*)d_in[11];
  const float* b_out  = (const float*)d_in[12];

  char* ws = (char*)d_ws;
  auto alloc = [&](size_t bytes) { char* q = ws; ws += (bytes + 255) & ~(size_t)255; return q; };
  float*  xp      = (float*) alloc((size_t)NPIX * C_  * 4);   // 25.2 MB
  float*  om      = (float*) alloc((size_t)NPIX * NOM * 4);   // 44.0 MB
  __bf16* x_bf    = (__bf16*)alloc((size_t)NPIX * C_  * 2);   // 12.6 MB
  __bf16* x1_bf   = (__bf16*)alloc((size_t)NPIX * C_  * 2);
  __bf16* y_bf    = (__bf16*)alloc((size_t)NPIX * C_  * 2);
  __bf16* w_inT   = (__bf16*)alloc((size_t)C_  * C_ * 2);
  __bf16* w_omT   = (__bf16*)alloc((size_t)NOM * C_ * 2);
  __bf16* w_outT  = (__bf16*)alloc((size_t)C_  * C_ * 2);
  float*  bias_om = (float*) alloc((size_t)NOM * 4);

  prep_weights<<<(NOM * C_ + 255) / 256, 256, 0, stream>>>(
      w_in, w_off, w_mask, w_out, b_off, b_mask, w_inT, w_omT, w_outT, bias_om);
  dw_ln_gelu<<<NPIX, C_, 0, stream>>>(x, dw_w, dw_b, ln_g, ln_b, x1_bf, x_bf);
  gemm_bf16<4><<<256 * 3, 256, 0, stream>>>(x_bf,  w_inT,  b_in,    xp,            192);
  gemm_bf16<3><<<256 * 7, 256, 0, stream>>>(x1_bf, w_omT,  bias_om, om,            336);
  deform_sample<<<(NPIX * G_) / 16, 256, 0, stream>>>(xp, om, y_bf);
  gemm_bf16<4><<<256 * 3, 256, 0, stream>>>(y_bf,  w_outT, b_out,   (float*)d_out, 192);
}
